// SenseMemAct_18588618457548
// MI455X (gfx1250) — matrix operand order repaired and checked by reading, not compiled, not run on hardware
//
#include <hip/hip_runtime.h>
#include <math.h>
#include <stdint.h>

typedef _Float16 v16h __attribute__((ext_vector_type(16)));
typedef _Float16 v8h  __attribute__((ext_vector_type(8)));
typedef float    v8f  __attribute__((ext_vector_type(8)));

#define BB   64
#define LL   4096
#define NIN  32
#define MEM  64

// workspace layout
#define WCOMB_OFF 0        // 192*32 f16 = 12 KB
#define XACO_OFF  32768    // BB*LL*192 f16 ~ 100.7 MB

#if __has_builtin(__builtin_amdgcn_tanhf)
#define TANHF(x) __builtin_amdgcn_tanhf(x)
#else
#define TANHF(x) tanhf(x)
#endif

// CDNA5 async copy: one wave-level instruction moves 32 lanes x 16B into LDS,
// tracked by ASYNCcnt (CDNA5 ISA). lds_off is the LDS
// byte offset (low 32 bits of the flat shared-aperture address).
__device__ __forceinline__ void async_g2l_b128(uint32_t lds_off, uint64_t gaddr) {
    asm volatile("global_load_async_to_lds_b128 %0, %1, off"
                 :: "v"(lds_off), "v"(gaddr) : "memory");
}
__device__ __forceinline__ void wait_asynccnt0() {
    asm volatile("s_wait_asynccnt 0x0" ::: "memory");
}

// ---------------------------------------------------------------------------
// Kernel 1: fold W_sense into the three input projections:
// Wcomb[m][n] = sum_i W_{ia|ic|io}[m][i] * W_sense[i][n], m in [0,192), n in [0,32)
// ---------------------------------------------------------------------------
__global__ void prep_wcomb(const float* __restrict__ Wsense,
                           const float* __restrict__ Wia,
                           const float* __restrict__ Wic,
                           const float* __restrict__ Wio,
                           _Float16* __restrict__ wcomb) {
    int tid = blockIdx.x * blockDim.x + threadIdx.x;
    if (tid >= 192 * 32) return;
    int m = tid >> 5, n = tid & 31;
    const float* src = (m < 64)  ? (Wia + m * 32)
                     : (m < 128) ? (Wic + (m - 64) * 32)
                                 : (Wio + (m - 128) * 32);
    float s = 0.f;
#pragma unroll
    for (int i = 0; i < 32; ++i) s += src[i] * Wsense[i * 32 + n];
    wcomb[m * 32 + n] = (_Float16)s;
}

// ---------------------------------------------------------------------------
// Kernel 2: XACO[t][b][0:192] = x[b,t,:] @ Wcomb^T   (one WMMA per 16x16 tile)
// Tile: fixed b, 16 consecutive timesteps (M), 16 output cols (N), K=32.
// ---------------------------------------------------------------------------
__global__ void __launch_bounds__(128) gemm_xaco(
    const float* __restrict__ x,
    const _Float16* __restrict__ wcomb,
    _Float16* __restrict__ xaco) {
    int wave = blockIdx.x * (blockDim.x >> 5) + (threadIdx.x >> 5);
    int lane = threadIdx.x & 31;
    int nt  = wave % 12;
    int tmp = wave / 12;
    int l0  = (tmp & 255) << 4;
    int b   = tmp >> 8;
    int g = lane >> 4, n = lane & 15;

    // A fragment: row m = lane&15 (timestep l0+m), 32 raw inputs
    const float* xr = x + ((size_t)b * LL + l0 + n) * NIN;
    v16h a;
#pragma unroll
    for (int i = 0; i < 8; ++i) {
        a[i]     = (_Float16)xr[16 * g + i];       // K = 16g + i
        a[i + 8] = (_Float16)xr[16 * g + 8 + i];   // K = 16g + 8 + i
    }
    // B fragment: lane = output col, halves -> K = 16g + i (contiguous)
    v16h bf = *(const v16h*)(wcomb + ((nt * 16 + n) * 32 + 16 * g));

    v8f acc = {};
    acc = __builtin_amdgcn_wmma_f32_16x16x32_f16(false, a, false, bf,
                                                 (short)0, acc, false, false);
#pragma unroll
    for (int j = 0; j < 8; ++j) {
        int m = j + 8 * g;                         // D: VGPR j -> row m, lane -> col
        xaco[((size_t)(l0 + m) * BB + b) * 192 + nt * 16 + n] = (_Float16)acc[j];
    }
}

// ---------------------------------------------------------------------------
// Kernel 3: recurrent scan. 4 blocks x 16 batches. Waves 0-3: WMMA recurrence,
// waves 4-5: fused readout+softmax of the previous step's H (pipelined).
// Per-step xa/xc/xo tile (6144 B contiguous) is double-buffered in LDS via
// ASYNCcnt-tracked global_load_async_to_lds_b128.
// ---------------------------------------------------------------------------
__global__ void __launch_bounds__(192) scan_kernel(
    const float* __restrict__ Wha, const float* __restrict__ Whc,
    const float* __restrict__ Wact, const _Float16* __restrict__ xaco,
    float* __restrict__ out) {
    __shared__ __align__(16) _Float16 H[2][16][64];      // hidden state, double buffer
    __shared__ __align__(16) _Float16 Xl[2][16 * 192];   // xa/xc/xo step tiles

    int tid = threadIdx.x;
    int b0  = blockIdx.x * 16;

    for (int i = tid; i < 2 * 16 * 64; i += blockDim.x)
        ((_Float16*)H)[i] = (_Float16)0.f;

    // async-stage step 0 tile into Xl[0] (192 thr x 2 x 16B = 6144 B)
    {
        uint64_t gbase = (uint64_t)(uintptr_t)xaco + ((size_t)b0 * 192) * 2;
        uint32_t lbase = (uint32_t)(uintptr_t)(&Xl[0][0]);
#pragma unroll
        for (int r = 0; r < 2; ++r) {
            uint32_t off = (uint32_t)tid * 32 + r * 16;
            async_g2l_b128(lbase + off, gbase + off);
        }
    }
    wait_asynccnt0();
    __syncthreads();

    bool is_compute = tid < 128;
    int lane = tid & 31;
    int g = lane >> 4, n = lane & 15;

    // ---- compute-wave state ----
    int ng = (tid >> 5) * 16 + n;        // global hidden column for this wave's tile
    v16h bha[2], bhc[2];
    float hprev[8];

    // ---- logits-wave state ----
    int q = tid - 128;
    int lm = q >> 2, ld = q & 3;
    bool lvalid = ld < 3;
    float wact[64];

    if (is_compute) {
#pragma unroll
        for (int c = 0; c < 2; ++c) {
            const float* pa = Wha + ng * 64 + 32 * c + 16 * g;   // B: K = 32c+16g+i
            const float* pc = Whc + ng * 64 + 32 * c + 16 * g;
#pragma unroll
            for (int i = 0; i < 16; ++i) {
                bha[c][i] = (_Float16)pa[i];
                bhc[c][i] = (_Float16)pc[i];
            }
        }
#pragma unroll
        for (int j = 0; j < 8; ++j) hprev[j] = 0.f;
    } else {
#pragma unroll
        for (int k = 0; k < 64; ++k) wact[k] = lvalid ? Wact[ld * 64 + k] : 0.f;
    }

    for (int t = 0; t < LL; ++t) {
        // issue async copy of step t+1 tile into the other buffer (all waves)
        {
            int tp1 = (t + 1 < LL) ? (t + 1) : t;
            uint64_t gbase = (uint64_t)(uintptr_t)xaco +
                             (((size_t)tp1 * BB + b0) * 192) * 2;
            uint32_t lbase = (uint32_t)(uintptr_t)(&Xl[(t + 1) & 1][0]);
#pragma unroll
            for (int r = 0; r < 2; ++r) {
                uint32_t off = (uint32_t)tid * 32 + r * 16;
                async_g2l_b128(lbase + off, gbase + off);
            }
        }

        if (is_compute) {
            const _Float16* xcur = &Xl[t & 1][0];
            // A fragments of h_{t-1} from LDS (buffer (t-1)&1 == (t+1)&1)
            int bi = (t + 1) & 1;
            v16h aa[2];
#pragma unroll
            for (int c = 0; c < 2; ++c) {
                v8h lo = *(const v8h*)(&H[bi][n][32 * c + 16 * g]);
                v8h hi = *(const v8h*)(&H[bi][n][32 * c + 16 * g + 8]);
                aa[c] = __builtin_shufflevector(lo, hi, 0, 1, 2, 3, 4, 5, 6, 7,
                                                8, 9, 10, 11, 12, 13, 14, 15);
            }
            v8f accA = {}, accC = {};
            accA = __builtin_amdgcn_wmma_f32_16x16x32_f16(false, aa[0], false, bha[0], (short)0, accA, false, false);
            accA = __builtin_amdgcn_wmma_f32_16x16x32_f16(false, aa[1], false, bha[1], (short)0, accA, false, false);
            accC = __builtin_amdgcn_wmma_f32_16x16x32_f16(false, aa[0], false, bhc[0], (short)0, accC, false, false);
            accC = __builtin_amdgcn_wmma_f32_16x16x32_f16(false, aa[1], false, bhc[1], (short)0, accC, false, false);
#pragma unroll
            for (int j = 0; j < 8; ++j) {
                int m = j + 8 * g;
                float xa = (float)xcur[m * 192 + ng];
                float xc = (float)xcur[m * 192 + 64 + ng];
                float xo = (float)xcur[m * 192 + 128 + ng];
                float av = 1.f + TANHF(accA[j] + xa);
                float cf = 1.f / (1.f + __expf(-(accC[j] + xc)));
                float hn = cf * hprev[j] + (1.f - cf) * TANHF(xo + av * hprev[j]);
                hprev[j] = hn;
                H[t & 1][m][ng] = (_Float16)hn;
            }
        } else if (t > 0) {
            // readout + softmax for step t-1 (buffer (t-1)&1)
            int bi = (t - 1) & 1;
            float s = 0.f;
#pragma unroll
            for (int c = 0; c < 8; ++c) {
                v8h hv = *(const v8h*)(&H[bi][lm][c * 8]);
#pragma unroll
                for (int i = 0; i < 8; ++i) s += (float)hv[i] * wact[c * 8 + i];
            }
            float lg = lvalid ? s : -1e30f;
            float mx = fmaxf(lg, __shfl_xor(lg, 1, 4));
            mx = fmaxf(mx, __shfl_xor(mx, 2, 4));
            float e = __expf(lg - mx);
            float sum = e + __shfl_xor(e, 1, 4);
            sum = sum + __shfl_xor(sum, 2, 4);
            if (lvalid)
                out[((size_t)(b0 + lm) * LL + (t - 1)) * 3 + ld] = e / sum;
        }

        wait_asynccnt0();     // t+1 tile landed in LDS
        __syncthreads();      // H[t] + Xl[(t+1)&1] visible to all waves
    }

    // epilogue: readout for t = LL-1
    if (!is_compute) {
        int bi = (LL - 1) & 1;
        float s = 0.f;
#pragma unroll
        for (int c = 0; c < 8; ++c) {
            v8h hv = *(const v8h*)(&H[bi][lm][c * 8]);
#pragma unroll
            for (int i = 0; i < 8; ++i) s += (float)hv[i] * wact[c * 8 + i];
        }
        float lg = lvalid ? s : -1e30f;
        float mx = fmaxf(lg, __shfl_xor(lg, 1, 4));
        mx = fmaxf(mx, __shfl_xor(mx, 2, 4));
        float e = __expf(lg - mx);
        float sum = e + __shfl_xor(e, 1, 4);
        sum = sum + __shfl_xor(sum, 2, 4);
        if (lvalid)
            out[((size_t)(b0 + lm) * LL + (LL - 1)) * 3 + ld] = e / sum;
    }
}

// ---------------------------------------------------------------------------
extern "C" void kernel_launch(void* const* d_in, const int* in_sizes, int n_in,
                              void* d_out, int out_size, void* d_ws, size_t ws_size,
                              hipStream_t stream) {
    const float* x      = (const float*)d_in[0];
    const float* Wsense = (const float*)d_in[1];
    const float* Wia    = (const float*)d_in[2];
    const float* Wha    = (const float*)d_in[3];
    const float* Wic    = (const float*)d_in[4];
    const float* Whc    = (const float*)d_in[5];
    const float* Wio    = (const float*)d_in[6];
    const float* Wact   = (const float*)d_in[7];
    float* out = (float*)d_out;

    _Float16* wcomb = (_Float16*)((char*)d_ws + WCOMB_OFF);
    _Float16* xaco  = (_Float16*)((char*)d_ws + XACO_OFF);

    prep_wcomb<<<24, 256, 0, stream>>>(Wsense, Wia, Wic, Wio, wcomb);

    int tiles = BB * (LL / 16) * 12;           // 196,608 independent WMMA tiles
    gemm_xaco<<<tiles / 4, 128, 0, stream>>>(x, wcomb, xaco);

    scan_kernel<<<BB / 16, 192, 0, stream>>>(Wha, Whc, Wact, xaco, out);
}
